// Attention_55207509622795
// MI455X (gfx1250) — matrix operand order repaired and checked by reading, not compiled, not run on hardware
//
#include <hip/hip_runtime.h>
#include <hip/hip_bf16.h>

typedef __bf16 bf16_t;
typedef __attribute__((ext_vector_type(16))) __bf16 v16bf;
typedef __attribute__((ext_vector_type(8)))  __bf16 v8bf;
typedef __attribute__((ext_vector_type(4)))  __bf16 v4bf;
typedef __attribute__((ext_vector_type(8)))  float  v8f;

#define LDSK 40   // padded LDS row (elements); 80B rows keep 16B alignment, spread banks
#define TILE_ELEMS (128 * LDSK)

// ---------------- WMMA helpers ----------------

__device__ __forceinline__ v8f wmma_bf16(v16bf a, v16bf b, v8f c) {
  return __builtin_amdgcn_wmma_f32_16x16x32_bf16(false, a, false, b, (short)0, c, false, false);
}

// A-fragment: lane (r,h): elements K = h*8+i (i<8) and 16+h*8+(i-8)
__device__ __forceinline__ v16bf frag_a(const bf16_t* rowp, int h) {
  v8bf lo = *reinterpret_cast<const v8bf*>(rowp + h * 8);
  v8bf hi = *reinterpret_cast<const v8bf*>(rowp + 16 + h * 8);
  v16bf f;
#pragma unroll
  for (int i = 0; i < 8; ++i) { f[i] = lo[i]; f[i + 8] = hi[i]; }
  return f;
}

// B-fragment (LDS stored [n][k]): lane (r,h): elements K = h*8+i (i<8) and 16+h*8+(i-8)
__device__ __forceinline__ v16bf frag_b(const bf16_t* rowp, int h) {
  v8bf lo = *reinterpret_cast<const v8bf*>(rowp + h * 8);
  v8bf hi = *reinterpret_cast<const v8bf*>(rowp + 16 + h * 8);
  v16bf f;
#pragma unroll
  for (int i = 0; i < 8; ++i) { f[i] = lo[i]; f[i + 8] = hi[i]; }
  return f;
}

// Wave (wm 0..3, wn 0..1) computes a 32x64 sub-tile: 2x4 grid of 16x16 WMMAs.
__device__ __forceinline__ void mma_step(const bf16_t* As, const bf16_t* Bs,
                                         int wm, int wn, int lane, v8f acc[2][4]) {
  const int r = lane & 15, h = lane >> 4;
  const v16bf a0 = frag_a(As + (wm * 32 +  0 + r) * LDSK, h);
  const v16bf a1 = frag_a(As + (wm * 32 + 16 + r) * LDSK, h);
#pragma unroll
  for (int nf = 0; nf < 4; ++nf) {
    const v16bf b = frag_b(Bs + (wn * 64 + nf * 16 + r) * LDSK, h);
    acc[0][nf] = wmma_bf16(a0, b, acc[0][nf]);
    acc[1][nf] = wmma_bf16(a1, b, acc[1][nf]);
  }
}

__device__ __forceinline__ void zero_acc(v8f acc[2][4]) {
#pragma unroll
  for (int i = 0; i < 2; ++i)
#pragma unroll
    for (int j = 0; j < 4; ++j)
#pragma unroll
      for (int e = 0; e < 8; ++e) acc[i][j][e] = 0.0f;
}

// Software-pipelined double-buffered GEMM: global->reg loads issue before the WMMA
// burst, reg->LDS stores land after it, so the s_wait_loadcnt hides under the WMMAs.
__device__ __forceinline__ void gemm_loop(const bf16_t* __restrict__ A, size_t lda,
                                          const bf16_t* __restrict__ B, size_t ldb,
                                          int KT, int tid, int lane, int wm, int wn,
                                          bf16_t (*As)[TILE_ELEMS], bf16_t (*Bs)[TILE_ELEMS],
                                          v8f acc[2][4]) {
  // per-thread addresses: thread covers rows (tid>>2) and (tid>>2)+64, 16B column chunk (tid&3)*8
  const int rowt = tid >> 2, colt = (tid & 3) * 8;
  const bf16_t* gA0 = A + (size_t)rowt * lda + colt;
  const bf16_t* gA1 = gA0 + (size_t)64 * lda;
  const bf16_t* gB0 = B + (size_t)rowt * ldb + colt;
  const bf16_t* gB1 = gB0 + (size_t)64 * ldb;
  const int loff0 = rowt * LDSK + colt;
  const int loff1 = loff0 + 64 * LDSK;

  uint4 ra0, ra1, rb0, rb1;
  ra0 = *reinterpret_cast<const uint4*>(gA0);
  ra1 = *reinterpret_cast<const uint4*>(gA1);
  rb0 = *reinterpret_cast<const uint4*>(gB0);
  rb1 = *reinterpret_cast<const uint4*>(gB1);
  gA0 += 32; gA1 += 32; gB0 += 32; gB1 += 32;
  *reinterpret_cast<uint4*>(As[0] + loff0) = ra0;
  *reinterpret_cast<uint4*>(As[0] + loff1) = ra1;
  *reinterpret_cast<uint4*>(Bs[0] + loff0) = rb0;
  *reinterpret_cast<uint4*>(Bs[0] + loff1) = rb1;
  __syncthreads();

  for (int kt = 0; kt < KT; kt += 2) {
    // stage slab kt+1 into registers (KT even => kt+1 always valid)
    ra0 = *reinterpret_cast<const uint4*>(gA0);
    ra1 = *reinterpret_cast<const uint4*>(gA1);
    rb0 = *reinterpret_cast<const uint4*>(gB0);
    rb1 = *reinterpret_cast<const uint4*>(gB1);
    gA0 += 32; gA1 += 32; gB0 += 32; gB1 += 32;
    mma_step(As[0], Bs[0], wm, wn, lane, acc);
    __builtin_amdgcn_sched_barrier(0);
    *reinterpret_cast<uint4*>(As[1] + loff0) = ra0;
    *reinterpret_cast<uint4*>(As[1] + loff1) = ra1;
    *reinterpret_cast<uint4*>(Bs[1] + loff0) = rb0;
    *reinterpret_cast<uint4*>(Bs[1] + loff1) = rb1;
    __syncthreads();

    const bool more = (kt + 2) < KT;
    if (more) {
      ra0 = *reinterpret_cast<const uint4*>(gA0);
      ra1 = *reinterpret_cast<const uint4*>(gA1);
      rb0 = *reinterpret_cast<const uint4*>(gB0);
      rb1 = *reinterpret_cast<const uint4*>(gB1);
      gA0 += 32; gA1 += 32; gB0 += 32; gB1 += 32;
    }
    mma_step(As[1], Bs[1], wm, wn, lane, acc);
    __builtin_amdgcn_sched_barrier(0);
    if (more) {
      *reinterpret_cast<uint4*>(As[0] + loff0) = ra0;
      *reinterpret_cast<uint4*>(As[0] + loff1) = ra1;
      *reinterpret_cast<uint4*>(Bs[0] + loff0) = rb0;
      *reinterpret_cast<uint4*>(Bs[0] + loff1) = rb1;
    }
    __syncthreads();
  }
}

// ---------------- Prep kernels ----------------

// W[3][k][n] f32  ->  Wt[3][n][k] bf16 (LDS-tiled transpose, run once)
__global__ void wt_kernel(const float* __restrict__ W, bf16_t* __restrict__ Wt) {
  __shared__ float t[32][33];
  const int z = blockIdx.z;
  const int k0 = blockIdx.y * 32, n0 = blockIdx.x * 32;
  const float* Wz = W + (size_t)z * 1024 * 1024;
  bf16_t* Wtz = Wt + (size_t)z * 1024 * 1024;
  const int tx = threadIdx.x & 31, ty = threadIdx.x >> 5;   // 32 x 8
#pragma unroll
  for (int r = 0; r < 32; r += 8)
    t[ty + r][tx] = Wz[(size_t)(k0 + ty + r) * 1024 + n0 + tx];
  __syncthreads();
#pragma unroll
  for (int r = 0; r < 32; r += 8)
    Wtz[(size_t)(n0 + ty + r) * 1024 + k0 + tx] = (bf16_t)t[tx][ty + r];
}

// x f32 -> bf16, elementwise (per batch)
__global__ void cvt_kernel(const float* __restrict__ in, bf16_t* __restrict__ out) {
  const size_t i = ((size_t)blockIdx.x * 256 + threadIdx.x) * 4;
  float4 v = *reinterpret_cast<const float4*>(in + i);
  v4bf p;
  p[0] = (bf16_t)v.x; p[1] = (bf16_t)v.y; p[2] = (bf16_t)v.z; p[3] = (bf16_t)v.w;
  *reinterpret_cast<v4bf*>(out + i) = p;
}

// ---------------- Kernel 1: QKV projection ----------------
__global__ void __launch_bounds__(256, 1)
qkv_kernel(const bf16_t* __restrict__ X16, const bf16_t* __restrict__ Wt,
           bf16_t* __restrict__ Q, bf16_t* __restrict__ K, bf16_t* __restrict__ Vt) {
  __shared__ bf16_t As[2][TILE_ELEMS];
  __shared__ bf16_t Bs[2][TILE_ELEMS];
  const int tid = threadIdx.x, lane = tid & 31, wid = tid >> 5;
  const int wm = wid >> 1, wn = wid & 1;
  const int n0 = blockIdx.x * 128, m0 = blockIdx.y * 128, z = blockIdx.z;

  v8f acc[2][4];
  zero_acc(acc);
  gemm_loop(X16 + (size_t)m0 * 1024, 1024,
            Wt + (size_t)z * 1024 * 1024 + (size_t)n0 * 1024, 1024,
            32, tid, lane, wm, wn, As, Bs, acc);

  const int r = lane & 15, h = lane >> 4;
  if (z == 2) {
    // store V transposed: Vt[n][m..m+7] -> one 16B store per tile per lane
#pragma unroll
    for (int mf = 0; mf < 2; ++mf)
#pragma unroll
      for (int nf = 0; nf < 4; ++nf) {
        int m = m0 + wm * 32 + mf * 16 + 8 * h;
        int n = n0 + wn * 64 + nf * 16 + r;
        v8bf p;
#pragma unroll
        for (int v = 0; v < 8; ++v) p[v] = (bf16_t)acc[mf][nf][v];
        *reinterpret_cast<v8bf*>(Vt + (size_t)n * 2048 + m) = p;
      }
  } else {
    bf16_t* outp = (z == 0) ? Q : K;
#pragma unroll
    for (int mf = 0; mf < 2; ++mf)
#pragma unroll
      for (int nf = 0; nf < 4; ++nf) {
        int m = m0 + wm * 32 + mf * 16 + 8 * h;
        int n = n0 + wn * 64 + nf * 16 + r;
#pragma unroll
        for (int v = 0; v < 8; ++v)
          outp[(size_t)(m + v) * 1024 + n] = (bf16_t)acc[mf][nf][v];
      }
  }
}

// ---------------- Kernel 2: scores = (Q @ K^T) * SCALE ----------------
__global__ void __launch_bounds__(256, 1)
scores_kernel(const bf16_t* __restrict__ Q, const bf16_t* __restrict__ Km,
              float* __restrict__ Sc) {
  __shared__ bf16_t As[2][TILE_ELEMS];
  __shared__ bf16_t Bs[2][TILE_ELEMS];
  const int tid = threadIdx.x, lane = tid & 31, wid = tid >> 5;
  const int wm = wid >> 1, wn = wid & 1;
  const int n0 = blockIdx.x * 128, m0 = blockIdx.y * 128;

  v8f acc[2][4];
  zero_acc(acc);
  gemm_loop(Q + (size_t)m0 * 1024, 1024,
            Km + (size_t)n0 * 1024, 1024,
            32, tid, lane, wm, wn, As, Bs, acc);

  const int r = lane & 15, h = lane >> 4;
#pragma unroll
  for (int mf = 0; mf < 2; ++mf)
#pragma unroll
    for (int nf = 0; nf < 4; ++nf) {
      int m = m0 + wm * 32 + mf * 16 + 8 * h;
      int n = n0 + wn * 64 + nf * 16 + r;
#pragma unroll
      for (int v = 0; v < 8; ++v)
        Sc[(size_t)(m + v) * 2048 + n] = acc[mf][nf][v] * 0.125f;  // 1/sqrt(64)
    }
}

// ---------------- Kernel 3: row softmax (f32 in, bf16 probs out) ----------------
__global__ void softmax_kernel(const float* __restrict__ Sc, bf16_t* __restrict__ P) {
  __shared__ float sm[256];
  const int q = blockIdx.x;
  const float* row = Sc + (size_t)q * 2048;
  float v[8];
  float m = -3.4e38f;
#pragma unroll
  for (int i = 0; i < 8; ++i) { v[i] = row[threadIdx.x + i * 256]; m = fmaxf(m, v[i]); }
  sm[threadIdx.x] = m; __syncthreads();
  for (int s = 128; s > 0; s >>= 1) {
    if (threadIdx.x < s) sm[threadIdx.x] = fmaxf(sm[threadIdx.x], sm[threadIdx.x + s]);
    __syncthreads();
  }
  m = sm[0]; __syncthreads();
  float sum = 0.0f;
#pragma unroll
  for (int i = 0; i < 8; ++i) { v[i] = __expf(v[i] - m); sum += v[i]; }
  sm[threadIdx.x] = sum; __syncthreads();
  for (int s = 128; s > 0; s >>= 1) {
    if (threadIdx.x < s) sm[threadIdx.x] += sm[threadIdx.x + s];
    __syncthreads();
  }
  const float inv = 1.0f / sm[0];
#pragma unroll
  for (int i = 0; i < 8; ++i)
    P[(size_t)q * 2048 + threadIdx.x + i * 256] = (bf16_t)(v[i] * inv);
}

// ---------------- Kernel 4: out = P @ V  (B read from transposed Vt[o][s]) ----------------
__global__ void __launch_bounds__(256, 1)
pv_kernel(const bf16_t* __restrict__ P, const bf16_t* __restrict__ Vt,
          float* __restrict__ O) {
  __shared__ bf16_t As[2][TILE_ELEMS];
  __shared__ bf16_t Bs[2][TILE_ELEMS];
  const int tid = threadIdx.x, lane = tid & 31, wid = tid >> 5;
  const int wm = wid >> 1, wn = wid & 1;
  const int n0 = blockIdx.x * 128, m0 = blockIdx.y * 128;

  v8f acc[2][4];
  zero_acc(acc);
  gemm_loop(P + (size_t)m0 * 2048, 2048,
            Vt + (size_t)n0 * 2048, 2048,
            64, tid, lane, wm, wn, As, Bs, acc);

  const int r = lane & 15, h = lane >> 4;
#pragma unroll
  for (int mf = 0; mf < 2; ++mf)
#pragma unroll
    for (int nf = 0; nf < 4; ++nf) {
      int m = m0 + wm * 32 + mf * 16 + 8 * h;
      int n = n0 + wn * 64 + nf * 16 + r;
#pragma unroll
      for (int v = 0; v < 8; ++v)
        O[(size_t)(m + v) * 1024 + n] = acc[mf][nf][v];
    }
}

// ---------------- Launch ----------------
extern "C" void kernel_launch(void* const* d_in, const int* in_sizes, int n_in,
                              void* d_out, int out_size, void* d_ws, size_t ws_size,
                              hipStream_t stream) {
  const float* x = (const float*)d_in[0];   // [8,2048,1024]
  const float* w = (const float*)d_in[1];   // [3,1024,1024]
  float* out = (float*)d_out;               // [8,2048,1024]
  char* ws = (char*)d_ws;

  const size_t MB = 1024 * 1024;
  bf16_t* Wt  = (bf16_t*)(ws);              //  6 MB: transposed bf16 weights
  bf16_t* X16 = (bf16_t*)(ws +  6 * MB);    //  4 MB: current batch x in bf16
  bf16_t* Qb  = (bf16_t*)(ws + 10 * MB);    //  4 MB
  bf16_t* Kb  = (bf16_t*)(ws + 14 * MB);    //  4 MB
  bf16_t* Vt  = (bf16_t*)(ws + 18 * MB);    //  4 MB (transposed [o][s])
  float*  Sb  = (float*) (ws + 22 * MB);    // 16 MB scores f32
  bf16_t* Pb  = (bf16_t*)(ws + 38 * MB);    //  8 MB probs bf16

  wt_kernel<<<dim3(32, 32, 3), 256, 0, stream>>>(w, Wt);

  for (int b = 0; b < 8; ++b) {
    const float* xb = x + (size_t)b * 2048 * 1024;
    cvt_kernel   <<<dim3(2048),     256, 0, stream>>>(xb, X16);
    qkv_kernel   <<<dim3(8, 16, 3), 256, 0, stream>>>(X16, Wt, Qb, Kb, Vt);
    scores_kernel<<<dim3(16, 16),   256, 0, stream>>>(Qb, Kb, Sb);
    softmax_kernel<<<dim3(2048),    256, 0, stream>>>(Sb, Pb);
    pv_kernel    <<<dim3(8, 16),    256, 0, stream>>>(Pb, Vt, out + (size_t)b * 2048 * 1024);
  }
}
